// PrototypeKmeansDivergence_47459388621131
// MI455X (gfx1250) — compile-verified, matrix operand order checked by reading, not run on hardware
//
#include <hip/hip_runtime.h>
#include <hip/hip_bf16.h>
#include <math.h>

typedef __attribute__((ext_vector_type(16))) _Float16 v16h;
typedef __attribute__((ext_vector_type(8)))  float    v8f;

#define BB 64
#define TT 512
#define DD 768
#define LL 4
#define KK 6
#define KM_ITERS 20
#define NCHUNK (DD / 32)   // 24 WMMA K-chunks for the assignment GEMM
#define QT 6               // N-tiles per wave in the one-hot GEMM (48/8)

__device__ __forceinline__ int ke_idx(int e, int lane) {
    // K offset of element e of a v16h A/B fragment (16x16x32 f16 layout)
    return ((e < 8) ? e : e + 8) + ((lane < 16) ? 0 : 8);
}

// out[m][d] = (1/counts[m]) * sum_{tags[t]==m} F[t][d]   for m < rowCap.
// tags[] is PADDED with 255 for t >= n, so the one-hot A matrix masks
// out-of-range points by itself: fragment builds are branch-free, full EXEC.
__device__ __forceinline__ void onehot_gemm(const float* __restrict__ F,
                                            const unsigned char* __restrict__ tags,
                                            int n, int rowCap,
                                            const int* __restrict__ counts,
                                            float* __restrict__ outLds,
                                            bool keepOnEmpty,
                                            int lane, int w)
{
    const int nKc  = (n + 31) >> 5;
    const int mrow = lane & 15;          // A-row (and B-column) for this lane
    v8f zero = {};
    v8f acc[QT];
    #pragma unroll
    for (int q = 0; q < QT; ++q) acc[q] = zero;

    const unsigned long long* tags64 = (const unsigned long long*)tags;

    for (int kc = 0; kc < nKc; ++kc) {
        const int kbase = kc * 32;
        // A fragment: two contiguous 8-byte tag runs per lane (ds_load_b64 x2)
        const int u = kc * 4 + ((lane < 16) ? 0 : 1);
        unsigned long long lo = tags64[u];
        unsigned long long hi = tags64[u + 2];
        v16h a;
        #pragma unroll
        for (int e = 0; e < 8; ++e) {
            a[e]     = (_Float16)(((int)((lo >> (8 * e)) & 0xFF) == mrow) ? 1.0f : 0.0f);
            a[e + 8] = (_Float16)(((int)((hi >> (8 * e)) & 0xFF) == mrow) ? 1.0f : 0.0f);
        }
        #pragma unroll
        for (int q = 0; q < QT; ++q) {
            const int ntb = (w * QT + q) * 16;
            v16h bf;
            #pragma unroll
            for (int e = 0; e < 16; ++e) {
                int t = kbase + ke_idx(e, lane);           // t < 512: always in-bounds
                bf[e] = (_Float16)F[(size_t)t * DD + ntb + mrow];
            }
            acc[q] = __builtin_amdgcn_wmma_f32_16x16x32_f16(
                         false, a, false, bf, (short)0, acc[q], false, false);
        }
    }
    // D-tile layout: VGPR r holds (m = r + (lane<16?0:8), col = lane&15)
    #pragma unroll
    for (int q = 0; q < QT; ++q) {
        const int ntb = (w * QT + q) * 16;
        #pragma unroll
        for (int r = 0; r < 8; ++r) {
            int m = r + ((lane < 16) ? 0 : 8);
            if (m < rowCap) {
                int c = counts[m];
                if (c > 0)               outLds[m * DD + ntb + mrow] = acc[q][r] / (float)c;
                else if (!keepOnEmpty)   outLds[m * DD + ntb + mrow] = 0.f;
            }
        }
    }
}

__global__ __launch_bounds__(256)
void proto_kmeans_kernel(const float* __restrict__ features,
                         const int*   __restrict__ lens,
                         const int*   __restrict__ labels,
                         const float* __restrict__ kprob,
                         float*       __restrict__ out)
{
    __shared__ float              cen[KK * DD];          // 18 KB  centers
    __shared__ float              cenNorm[8];
    __shared__ _Float16           bStage[NCHUNK * 512];  // 24 KB  assignment B fragments
    __shared__ float              dTile[8 * 256];        //  8 KB  per-wave 16x16 D tiles
    __shared__ unsigned long long assignArr64[TT / 8];   // 0.5 KB (8-byte aligned tags)
    __shared__ unsigned long long labLds64[TT / 8];      // 0.5 KB
    __shared__ float              proto[LL * DD];        // 12 KB  prototypes
    __shared__ int                pCount[16];
    __shared__ int                cCount[16];
    __shared__ int                sLR;
    __shared__ int                sK;

    unsigned char* assignArr = (unsigned char*)assignArr64;
    unsigned char* labLds    = (unsigned char*)labLds64;

    const int b    = blockIdx.x;
    const int tid  = threadIdx.x;
    const int lane = tid & 31;
    const int w    = tid >> 5;
    const float* F   = features + (size_t)b * TT * DD;
    const int*   Lab = labels   + (size_t)b * TT;
    const int    n   = lens[b];

    // warm L2 for this batch's feature block (global_prefetch_b8)
    __builtin_prefetch(F + (size_t)tid * DD / 2, 0, 1);

    // ---- tags to LDS (padded with 255), counts, label_range, k ----
    if (tid < 16) { pCount[tid] = 0; cCount[tid] = 0; }
    if (tid == 0) {
        sLR = 1;
        int bj = 0; float bv = kprob[b * KK];
        for (int j = 1; j < KK; ++j) {
            float v = kprob[b * KK + j];
            if (v > bv) { bv = v; bj = j; }
        }
        sK = bj + 1;
    }
    for (int t = tid; t < TT; t += 256) {
        labLds[t]    = (t < n) ? (unsigned char)Lab[t] : (unsigned char)255;
        assignArr[t] = (unsigned char)255;
    }
    __syncthreads();
    for (int t = tid; t < n; t += 256) {
        int l = labLds[t];
        atomicAdd(&pCount[l], 1);
        atomicMax(&sLR, l + 1);
    }
    __syncthreads();
    const int LR = sLR;
    const int k  = sK;

    // ---- prototypes = onehot(labels)^T * X  via WMMA ----
    onehot_gemm(F, labLds, n, LR, pCount, proto, /*keepOnEmpty=*/false, lane, w);

    // ---- deterministic spread init of centers ----
    for (int idx = tid; idx < k * DD; idx += 256) {
        int j = idx / DD, d = idx % DD;
        int t = (int)(((long long)j * n) / k);
        cen[j * DD + d] = F[(size_t)t * DD + d];
    }
    __syncthreads();

    const int nT = (n + 15) >> 4;   // 16-point tiles

    for (int it = 0; it < KM_ITERS; ++it) {
        // center norms
        if (tid < k) {
            float s = 0.f;
            for (int d = 0; d < DD; ++d) { float c = cen[tid * DD + d]; s += c * c; }
            cenNorm[tid] = s;
        }
        // stage centers as f16 B fragments (N padded to 16 with zeros)
        for (int idx = tid; idx < NCHUNK * 512; idx += 256) {
            int chunk = idx >> 9;
            int r     = idx & 511;
            int L2    = r >> 4;
            int e     = r & 15;
            int N     = L2 & 15;
            int Kd    = chunk * 32 + ke_idx(e, L2);
            float v   = (N < k) ? cen[N * DD + Kd] : 0.f;
            bStage[idx] = (_Float16)v;
        }
        __syncthreads();

        // ---- assignment: argmin_j ||c_j||^2 - 2 x.c_j  via WMMA ----
        for (int tile = w; tile < nT; tile += 8) {
            const float* row = F + (size_t)(tile * 16 + (lane & 15)) * DD;
            v8f acc = {};
            for (int c = 0; c < NCHUNK; ++c) {
                v16h a, bf;
                #pragma unroll
                for (int e = 0; e < 16; ++e)
                    a[e] = (_Float16)row[c * 32 + ke_idx(e, lane)];
                const _Float16* bp = &bStage[c * 512 + lane * 16];
                #pragma unroll
                for (int e = 0; e < 16; ++e) bf[e] = bp[e];
                acc = __builtin_amdgcn_wmma_f32_16x16x32_f16(
                          false, a, false, bf, (short)0, acc, false, false);
            }
            float* dz = &dTile[w * 256];
            #pragma unroll
            for (int r = 0; r < 8; ++r) {
                int m = r + ((lane < 16) ? 0 : 8);
                dz[m * 16 + (lane & 15)] = acc[r];
            }
            asm volatile("s_wait_dscnt 0" ::: "memory");
            if (lane < 16) {
                int t = tile * 16 + lane;
                float bv = 1e30f; int bj = 0;
                for (int j = 0; j < k; ++j) {
                    float v = cenNorm[j] - 2.f * dz[lane * 16 + j];
                    if (v < bv) { bv = v; bj = j; }
                }
                assignArr[t] = (t < n) ? (unsigned char)bj : (unsigned char)255;
            }
        }
        __syncthreads();

        // ---- cluster counts ----
        if (tid < 16) cCount[tid] = 0;
        __syncthreads();
        for (int t = tid; t < n; t += 256) atomicAdd(&cCount[assignArr[t]], 1);
        __syncthreads();

        // ---- center update = onehot(assign)^T * X  via WMMA ----
        onehot_gemm(F, assignArr, n, k, cCount, cen, /*keepOnEmpty=*/true, lane, w);
        __syncthreads();
    }

    // ---- cost matrix + exhaustive optimal assignment + loss ----
    if (tid == 0) {
        float cost[LL][KK];
        for (int r = 0; r < LR; ++r)
            for (int c = 0; c < k; ++c) {
                float s = 0.f;
                for (int d = 0; d < DD; ++d) {
                    float df = proto[r * DD + d] - cen[c * DD + d];
                    s += df * df;
                }
                cost[r][c] = sqrtf(s);
            }
        int  small    = (LR < k) ? LR : k;
        int  large    = (LR < k) ? k  : LR;
        bool rowSmall = (LR <= k);
        int lim1 = (small > 1) ? large : 1;
        int lim2 = (small > 2) ? large : 1;
        int lim3 = (small > 3) ? large : 1;
        float best = 1e30f;
        for (int i0 = 0; i0 < large; ++i0)
        for (int i1 = 0; i1 < lim1; ++i1) {
            if (small > 1 && i1 == i0) continue;
            for (int i2 = 0; i2 < lim2; ++i2) {
                if (small > 2 && (i2 == i0 || i2 == i1)) continue;
                for (int i3 = 0; i3 < lim3; ++i3) {
                    if (small > 3 && (i3 == i0 || i3 == i1 || i3 == i2)) continue;
                    float s = rowSmall ? cost[0][i0] : cost[i0][0];
                    if (small > 1) s += rowSmall ? cost[1][i1] : cost[i1][1];
                    if (small > 2) s += rowSmall ? cost[2][i2] : cost[i2][2];
                    if (small > 3) s += rowSmall ? cost[3][i3] : cost[i3][3];
                    if (s < best) best = s;
                }
            }
        }
        float meanDist = best / (float)small;
        float loss = 0.5f * meanDist - 0.5f * logf(kprob[b * KK + (LR - 1)]);
        atomicAdd(out, loss * (1.0f / (float)BB));
    }
}

extern "C" void kernel_launch(void* const* d_in, const int* in_sizes, int n_in,
                              void* d_out, int out_size, void* d_ws, size_t ws_size,
                              hipStream_t stream)
{
    (void)in_sizes; (void)n_in; (void)out_size; (void)d_ws; (void)ws_size;
    const float* features = (const float*)d_in[0];
    const int*   lens     = (const int*)d_in[1];
    const int*   labels   = (const int*)d_in[2];
    const float* kprob    = (const float*)d_in[3];
    float*       out      = (float*)d_out;

    hipMemsetAsync(out, 0, sizeof(float), stream);
    hipLaunchKernelGGL(proto_kmeans_kernel, dim3(BB), dim3(256), 0, stream,
                       features, lens, labels, kprob, out);
}
